// RCDNet_5549097747123
// MI455X (gfx1250) — matrix operand order repaired and checked by reading, not compiled, not run on hardware
//
#include <hip/hip_runtime.h>

typedef _Float16 h16;
typedef __attribute__((ext_vector_type(16))) _Float16 v16h;
typedef __attribute__((ext_vector_type(8)))  float    v8f;

#define LDC 144   // padded output width: 128 message cols + denom/score cols + pad
#define NT  9     // 144 / 16 N-tiles, all owned by one wave

template<int O>
__device__ __forceinline__ void put_mask4(v16h& a, int4 c) {
    a[O + 0] = (h16)(c.x > 0 ? 1.f : 0.f);
    a[O + 1] = (h16)(c.y > 0 ? 1.f : 0.f);
    a[O + 2] = (h16)(c.z > 0 ? 1.f : 0.f);
    a[O + 3] = (h16)(c.w > 0 ? 1.f : 0.f);
}
template<int O>
__device__ __forceinline__ void put_f4(v16h& a, float4 c) {
    a[O + 0] = (h16)c.x; a[O + 1] = (h16)c.y;
    a[O + 2] = (h16)c.z; a[O + 3] = (h16)c.w;
}

// ---------------------------------------------------------------------------
// Unified WMMA GEMM:  C[M x 144] = A[M x K] @ B[K x 144]
//   One wave per block owns a 16-row strip and all 9 N-tiles (9 accumulators),
//   so the A fragment is loaded once per k-step and reused 9x. No LDS, no
//   barriers.  B is stored column-major (BT[144][ldk], f16, 32B-aligned cols)
//   so each lane's B fragment is two contiguous 16B vector loads.
//   A source (template):
//     0: binary mask, row-major int32 indicator  A[m][k] = ind[m*lda+k] > 0
//     1: binary mask, transposed                 A[m][k] = ind[k*lda+m] > 0
//     2: q-masked raw score                      A[m][k] = q[k*128+m]>0 ? rk[m]+ri[k] : 0
//     3: fp32 matrix                             A[m][k] = Af[m*lda+k]
// ---------------------------------------------------------------------------
template<int ASRC>
__global__ __launch_bounds__(32)
void gemm_wmma(const int* __restrict__ Ai, const float* __restrict__ Af,
               const float* __restrict__ rk, const float* __restrict__ ri,
               const h16* __restrict__ BT, int ldk,
               float* __restrict__ C, int M, int K, int lda)
{
    const int r0    = blockIdx.x * 16;
    const int lane  = threadIdx.x & 31;
    const int half  = lane >> 4;
    const int arow  = lane & 15;
    const int abase = half << 3;     // A: lanes 16-31 hold K = 8..15 / 24..31
    const int koff  = half << 4;     // unused; B lanes 16-31 also hold K = 8..15 / 24..31 (abase)
    const int rowc  = (r0 + arow < M) ? (r0 + arow) : (M - 1);   // clamp (OOB rows not stored)

    const int*   __restrict__ api = (ASRC == 0) ? (Ai + (long)rowc * lda) : nullptr;
    const float* __restrict__ apf = (ASRC == 3) ? (Af + (long)rowc * lda) : nullptr;
    float rkm = 0.f;
    if (ASRC == 2) rkm = rk[rowc];

    // scalar A-element fetch (used by strided variants and the K tail)
    auto aval = [&](int gk) -> float {
        if (ASRC == 0) return (api[gk] > 0) ? 1.f : 0.f;
        if (ASRC == 1) return (Ai[(long)gk * lda + rowc] > 0) ? 1.f : 0.f;
        if (ASRC == 2) return (Ai[(long)gk * 128 + rowc] > 0) ? (rkm + ri[gk]) : 0.f;
        return apf[gk];
    };

    v8f acc[NT] = {};

    const int kmain = K & ~31;
    for (int k0 = 0; k0 < kmain; k0 += 32) {
        if (ASRC == 0) __builtin_prefetch(api + k0 + 512, 0, 1);
        if (ASRC == 1) __builtin_prefetch(Ai + (long)(k0 + 64) * lda + rowc, 0, 1);

        v16h a;
        if (ASRC == 0) {           // contiguous 16B chunks -> b128 loads
            put_mask4<0 >(a, *(const int4*)(api + k0 + abase));
            put_mask4<4 >(a, *(const int4*)(api + k0 + abase + 4));
            put_mask4<8 >(a, *(const int4*)(api + k0 + 16 + abase));
            put_mask4<12>(a, *(const int4*)(api + k0 + 16 + abase + 4));
        } else if (ASRC == 3) {
            put_f4<0 >(a, *(const float4*)(apf + k0 + abase));
            put_f4<4 >(a, *(const float4*)(apf + k0 + abase + 4));
            put_f4<8 >(a, *(const float4*)(apf + k0 + 16 + abase));
            put_f4<12>(a, *(const float4*)(apf + k0 + 16 + abase + 4));
        } else {                   // strided reads, 64B-coalesced across lanes
            #pragma unroll
            for (int j = 0; j < 8; ++j) {
                a[j]     = (h16)aval(k0 + abase + j);
                a[8 + j] = (h16)aval(k0 + 16 + abase + j);
            }
        }

        #pragma unroll
        for (int t = 0; t < NT; ++t) {
            typedef __attribute__((ext_vector_type(8))) _Float16 v8h_t;
            const h16* bq = BT + (long)(t * 16 + arow) * ldk + k0 + abase;
            const v8h_t blo = *(const v8h_t*)bq;
            const v8h_t bhi = *(const v8h_t*)(bq + 16);
            const v16h b = __builtin_shufflevector(blo, bhi, 0, 1, 2, 3, 4, 5, 6, 7, 8, 9, 10, 11, 12, 13, 14, 15);
            acc[t] = __builtin_amdgcn_wmma_f32_16x16x32_f16(false, a, false, b,
                                                            (short)0, acc[t], false, false);
        }
    }

    if (kmain < K) {               // single guarded tail step
        const int k0 = kmain;
        v16h a;
        #pragma unroll
        for (int j = 0; j < 8; ++j) {
            const int g1 = k0 + abase + j, g2 = k0 + 16 + abase + j;
            a[j]     = (h16)((g1 < K) ? aval(g1) : 0.f);
            a[8 + j] = (h16)((g2 < K) ? aval(g2) : 0.f);
        }
        #pragma unroll
        for (int t = 0; t < NT; ++t) {
            const h16* bp = BT + (long)(t * 16 + arow) * ldk + k0;
            v16h b;
            #pragma unroll
            for (int j = 0; j < 16; ++j) { const int kk = (j < 8) ? (abase + j) : (8 + abase + j); b[j] = (k0 + kk < K) ? bp[kk] : (h16)0.f; }
            acc[t] = __builtin_amdgcn_wmma_f32_16x16x32_f16(false, a, false, b,
                                                            (short)0, acc[t], false, false);
        }
    }

    // store (f32 C layout: VGPR v -> M = v / v+8, N = lane&15)
    #pragma unroll
    for (int t = 0; t < NT; ++t) {
        #pragma unroll
        for (int v = 0; v < 8; ++v) {
            const int m = r0 + (half << 3) + v;
            if (m < M) C[(long)m * LDC + t * 16 + arow] = acc[t][v];
        }
    }
}

// ---------------------------------------------------------------------------
// Build the 4 extended weight matrices, column-major f16.
// Extra columns fold the attention vectors a into W (score = emb @ (W @ a_part)).
// ---------------------------------------------------------------------------
__global__ __launch_bounds__(128)
void prep_bt(const float* __restrict__ Wst, const float* __restrict__ Wits,
             const float* __restrict__ Wik, const float* __restrict__ Wki,
             const float* __restrict__ a_stu_item, const float* __restrict__ a_item_stu,
             const float* __restrict__ a_item_skill, const float* __restrict__ a_skill_item,
             h16* __restrict__ BT1, h16* __restrict__ BT2,
             h16* __restrict__ BT3, h16* __restrict__ BT4)
{
    const int k = threadIdx.x;      // 0..127
    const int n = blockIdx.x;       // 0..143
    const int mat = blockIdx.y;     // 0..3
    float v = 0.f;
    if (mat == 0) {
        if (n < 128) v = Wits[k * 128 + n];
        else if (n == 128) { for (int j = 0; j < 128; ++j) v += Wits[k * 128 + j] * a_item_stu[128 + j]; }
    } else if (mat == 1) {
        if (n < 128) v = Wst[k * 128 + n];
        else if (n == 128) { for (int j = 0; j < 128; ++j) v += Wst[k * 128 + j] * a_stu_item[128 + j]; }
        else if (n == 129) { for (int j = 0; j < 128; ++j) v += Wki[k * 128 + j] * a_skill_item[128 + j]; }
    } else if (mat == 2) {
        if (n < 128) v = Wki[k * 128 + n];
    } else {
        if (n < 128) v = Wik[k * 128 + n];
        else if (n == 128) { for (int j = 0; j < 128; ++j) v += Wik[k * 128 + j] * a_item_skill[128 + j]; }
        else if (n == 129) { for (int j = 0; j < 128; ++j) v += Wki[k * 128 + j] * a_skill_item[j]; }
    }
    h16* BT = (mat == 0) ? BT1 : (mat == 1) ? BT2 : (mat == 2) ? BT3 : BT4;
    BT[n * 128 + k] = (h16)v;
}

// scal[0] = max_i s_i (Out2 col 128), scal[1] = max_u t_u (Out1 col 128)
__global__ __launch_bounds__(256)
void reduce_max2(const float* __restrict__ Out2, const float* __restrict__ Out1,
                 float* __restrict__ scal)
{
    __shared__ float red[256];
    const float* src = blockIdx.x ? Out1 : Out2;
    const int n = blockIdx.x ? 8000 : 5000;
    float mx = -1e30f;
    for (int i = threadIdx.x; i < n; i += 256) mx = fmaxf(mx, src[(long)i * LDC + 128]);
    red[threadIdx.x] = mx; __syncthreads();
    for (int s = 128; s > 0; s >>= 1) {
        if (threadIdx.x < s) red[threadIdx.x] = fmaxf(red[threadIdx.x], red[threadIdx.x + s]);
        __syncthreads();
    }
    if (threadIdx.x == 0) scal[blockIdx.x] = red[0];
}

__global__ __launch_bounds__(256)
void extract_rvec(const float* __restrict__ Out2, const float* __restrict__ Out4,
                  float* __restrict__ ri, float* __restrict__ rkv)
{
    const int i = blockIdx.x * 256 + threadIdx.x;
    if (i < 5000) ri[i] = Out2[(long)i * LDC + 129];
    if (i < 128)  rkv[i] = Out4[(long)i * LDC + 129];
}

// Build column-major f16 B operands for the big masked GEMMs.
// mode 0: X1T[n][i] = exp(s_i - smax) * (n<128 ? Ti[i][n] : (n==128 ? 1 : 0))
// mode 1: X2T[n][u] = exp(t_u - tmax) * (n<128 ? Tis_u[u][n] : (n==128 ? 1 : 0))
// mode 2: X3T[n][i] = n<128 ? Tki_i[i][n] : 0
__global__ __launch_bounds__(256)
void xprep(const float* __restrict__ Out, const float* __restrict__ scal,
           h16* __restrict__ XT, int K, int ldk, int mode)
{
    const int n = blockIdx.x;                       // 0..143
    const int i = blockIdx.y * 256 + threadIdx.x;
    if (i >= ldk) return;
    float v = 0.f;
    if (i < K) {
        if (mode == 2) { if (n < 128) v = Out[(long)i * LDC + n]; }
        else {
            const float w = __expf(Out[(long)i * LDC + 128] - scal[mode]);
            if (n < 128)       v = w * Out[(long)i * LDC + n];
            else if (n == 128) v = w;
        }
    }
    XT[(long)n * ldk + i] = (h16)v;
}

// fu[u][d] = user_emb + C1[u][d] / den   (den = C1 col 128; empty rows -> 0)
__global__ __launch_bounds__(256)
void user_epilogue(const float* __restrict__ C1, const float* __restrict__ user_emb,
                   float* __restrict__ fu)
{
    const int idx = blockIdx.x * 256 + threadIdx.x;
    if (idx >= 8000 * 128) return;
    const int u = idx >> 7, d = idx & 127;
    const float den = C1[(long)u * LDC + 128];
    fu[idx] = user_emb[idx] + C1[(long)u * LDC + d] / fmaxf(den, 1e-30f);
}

// Per item: stu_msg from C2, skill softmax over 128 skills, map gates, item_out.
__global__ __launch_bounds__(128)
void item_epilogue(const float* __restrict__ C2, const float* __restrict__ item_emb,
                   const float* __restrict__ Out4, const int* __restrict__ q,
                   const float* __restrict__ ams, const float* __restrict__ amk,
                   float* __restrict__ fi)
{
    __shared__ float e[128];
    __shared__ float red[128];
    const int i = blockIdx.x, d = threadIdx.x;

    const float den2 = C2[(long)i * LDC + 128];
    const float smd  = C2[(long)i * LDC + d] / fmaxf(den2, 1e-30f);   // stu_msg[d]

    const int   msk = q[(long)i * 128 + d] > 0;
    const float pk  = Out4[(long)d * LDC + 128];
    red[d] = msk ? pk : -1e30f;
    __syncthreads();
    for (int s = 64; s > 0; s >>= 1) { if (d < s) red[d] = fmaxf(red[d], red[d + s]); __syncthreads(); }
    const float mx = red[0]; __syncthreads();
    const float ev = msk ? __expf(pk - mx) : 0.f;
    e[d] = ev; red[d] = ev;
    __syncthreads();
    for (int s = 64; s > 0; s >>= 1) { if (d < s) red[d] += red[d + s]; __syncthreads(); }
    const float den = fmaxf(red[0], 1e-12f); __syncthreads();

    float acc = 0.f;
    for (int k = 0; k < 128; ++k) acc += e[k] * Out4[(long)k * LDC + d];   // Tik_k[k][d]
    const float skd = acc / den;                                            // skill_msg[d]

    const float ie = item_emb[(long)i * 128 + d];
    red[d] = ie * ams[d] + smd * ams[128 + d];
    __syncthreads();
    for (int s = 64; s > 0; s >>= 1) { if (d < s) red[d] += red[d + s]; __syncthreads(); }
    const float map_stu = red[0]; __syncthreads();
    red[d] = ie * amk[d] + skd * amk[128 + d];
    __syncthreads();
    for (int s = 64; s > 0; s >>= 1) { if (d < s) red[d] += red[d + s]; __syncthreads(); }
    const float map_skill = red[0];

    fi[(long)i * 128 + d] = ie + map_stu * smd + map_skill * skd;
}

__global__ __launch_bounds__(256)
void skill_epilogue(const float* __restrict__ CD, const float* __restrict__ skill_emb,
                    float* __restrict__ fk)
{
    const int idx = blockIdx.x * 256 + threadIdx.x;
    if (idx >= 128 * 128) return;
    const int m = idx >> 7, d = idx & 127;
    fk[idx] = skill_emb[idx] + CD[(long)m * LDC + d];
}

// Prediction head: one block per batch element, 128 threads (one per feature).
__global__ __launch_bounds__(128)
void head_kernel(const int* __restrict__ user, const int* __restrict__ item,
                 const int* __restrict__ q,
                 const float* __restrict__ fu, const float* __restrict__ fi,
                 const float* __restrict__ fk,
                 const float* __restrict__ Wfs, const float* __restrict__ bfs,
                 const float* __restrict__ Wfi, const float* __restrict__ bfi,
                 const float* __restrict__ Wp, const float* __restrict__ bp,
                 float* __restrict__ out)
{
    __shared__ float su[128], sv[128], sa[128], red[128];
    const int b = blockIdx.x, d = threadIdx.x;
    const int u = user[b], it = item[b];
    su[d] = fu[(long)u * 128 + d];
    sv[d] = fi[(long)it * 128 + d];
    float acc = 0.f, qs = 0.f;
    for (int k = 0; k < 128; ++k) {
        const float qv = (float)q[(long)it * 128 + k];
        acc += qv * fk[(long)k * 128 + d];
        qs  += qv;
    }
    sa[d] = acc / fmaxf(qs, 1.f);
    __syncthreads();
    float s1 = bfs[d], s2 = bfi[d];
    for (int j = 0; j < 128; ++j) {
        s1 += su[j] * Wfs[(long)j * 128 + d];
        s2 += sv[j] * Wfi[(long)j * 128 + d];
    }
    for (int j = 0; j < 128; ++j) {
        const float c = sa[j];
        s1 += c * Wfs[(long)(128 + j) * 128 + d];
        s2 += c * Wfi[(long)(128 + j) * 128 + d];
    }
    const float f1 = 1.f / (1.f + __expf(-s1));
    const float f2 = 1.f / (1.f + __expf(-s2));
    red[d] = (f1 - f2) * Wp[d];
    __syncthreads();
    for (int s = 64; s > 0; s >>= 1) { if (d < s) red[d] += red[d + s]; __syncthreads(); }
    if (d == 0) out[b] = 1.f / (1.f + __expf(-(red[0] + bp[0])));
}

// ---------------------------------------------------------------------------
extern "C" void kernel_launch(void* const* d_in, const int* in_sizes, int n_in,
                              void* d_out, int out_size, void* d_ws, size_t ws_size,
                              hipStream_t stream)
{
    const int*   user      = (const int*)  d_in[0];
    const int*   item      = (const int*)  d_in[1];
    const int*   q         = (const int*)  d_in[2];
    const int*   indicator = (const int*)  d_in[3];
    const float* user_emb  = (const float*)d_in[4];
    const float* item_emb  = (const float*)d_in[5];
    const float* skill_emb = (const float*)d_in[6];
    const float* W_stu_T   = (const float*)d_in[7];
    const float* a_stu_it  = (const float*)d_in[8];
    const float* W_it_stu  = (const float*)d_in[9];
    const float* a_it_stu  = (const float*)d_in[10];
    const float* W_it_sk   = (const float*)d_in[11];
    const float* a_it_sk   = (const float*)d_in[12];
    const float* a_map_stu = (const float*)d_in[13];
    const float* a_map_sk  = (const float*)d_in[14];
    const float* W_sk_it   = (const float*)d_in[15];
    const float* a_sk_it   = (const float*)d_in[16];
    const float* W_fuse_s  = (const float*)d_in[17];
    const float* b_fuse_s  = (const float*)d_in[18];
    const float* W_fuse_i  = (const float*)d_in[19];
    const float* b_fuse_i  = (const float*)d_in[20];
    const float* W_pred    = (const float*)d_in[21];
    const float* b_pred    = (const float*)d_in[22];
    float* out = (float*)d_out;

    char* ws = (char*)d_ws;
    size_t off = 0;
    auto alloc = [&](size_t bytes) { char* p = ws + off; off = (off + bytes + 255) & ~(size_t)255; return p; };

    float* Out1 = (float*)alloc((size_t)8000 * LDC * 4);   // Tis_u | t_u      (reused as C1)
    float* Out2 = (float*)alloc((size_t)5000 * LDC * 4);   // Ti | s_i | r_i   (reused as C2)
    float* Out3 = (float*)alloc((size_t)5000 * LDC * 4);   // Tki_i            (reused as CD)
    float* Out4 = (float*)alloc((size_t)128  * LDC * 4);   // Tik_k | p_k | r_k
    h16* BT1 = (h16*)alloc((size_t)LDC * 128 * 2);
    h16* BT2 = (h16*)alloc((size_t)LDC * 128 * 2);
    h16* BT3 = (h16*)alloc((size_t)LDC * 128 * 2);
    h16* BT4 = (h16*)alloc((size_t)LDC * 128 * 2);
    h16* X1T = (h16*)alloc((size_t)LDC * 5008 * 2);
    h16* X2T = (h16*)alloc((size_t)LDC * 8000 * 2);
    h16* X3T = (h16*)alloc((size_t)LDC * 5008 * 2);
    float* scal = (float*)alloc(2 * 4);
    float* ri   = (float*)alloc(5000 * 4);
    float* rkv  = (float*)alloc(128 * 4);
    float* fu   = (float*)alloc((size_t)8000 * 128 * 4);
    float* fi   = (float*)alloc((size_t)5000 * 128 * 4);
    float* fk   = (float*)alloc((size_t)128  * 128 * 4);
    (void)ws_size; (void)n_in; (void)in_sizes; (void)out_size;

    // Stage A: extended weights + embedding projections (WMMA)
    prep_bt<<<dim3(LDC, 4), 128, 0, stream>>>(W_stu_T, W_it_stu, W_it_sk, W_sk_it,
                                              a_stu_it, a_it_stu, a_it_sk, a_sk_it,
                                              BT1, BT2, BT3, BT4);
    gemm_wmma<3><<<500, 32, 0, stream>>>(nullptr, user_emb,  nullptr, nullptr, BT1, 128, Out1, 8000, 128, 128);
    gemm_wmma<3><<<313, 32, 0, stream>>>(nullptr, item_emb,  nullptr, nullptr, BT2, 128, Out2, 5000, 128, 128);
    gemm_wmma<3><<<313, 32, 0, stream>>>(nullptr, item_emb,  nullptr, nullptr, BT3, 128, Out3, 5000, 128, 128);
    gemm_wmma<3><<<  8, 32, 0, stream>>>(nullptr, skill_emb, nullptr, nullptr, BT4, 128, Out4,  128, 128, 128);

    // Stage B prep: stabilizing maxima, compact score vectors, f16 B operands
    reduce_max2<<<2, 256, 0, stream>>>(Out2, Out1, scal);
    extract_rvec<<<20, 256, 0, stream>>>(Out2, Out4, ri, rkv);
    xprep<<<dim3(LDC, 20), 256, 0, stream>>>(Out2, scal, X1T, 5000, 5008, 0);
    xprep<<<dim3(LDC, 32), 256, 0, stream>>>(Out1, scal, X2T, 8000, 8000, 1);
    xprep<<<dim3(LDC, 20), 256, 0, stream>>>(Out3, scal, X3T, 5000, 5008, 2);

    // Stage B: the three mask GEMMs (WMMA; C aliases the dead Out buffers)
    float* C1 = Out1;   // [8000 x 144]: cols 0-127 = M @ (w .* Ti), col 128 = M @ w
    float* C2 = Out2;   // [5000 x 144]: M^T variant
    float* CD = Out3;   // [128  x 144]: raw-score skill aggregation
    gemm_wmma<0><<<500, 32, 0, stream>>>(indicator, nullptr, nullptr, nullptr, X1T, 5008, C1, 8000, 5000, 5000);
    gemm_wmma<1><<<313, 32, 0, stream>>>(indicator, nullptr, nullptr, nullptr, X2T, 8000, C2, 5000, 8000, 5000);
    gemm_wmma<2><<<  8, 32, 0, stream>>>(q,         nullptr, rkv,     ri,      X3T, 5008, CD,  128, 5000,  128);

    // Epilogues + prediction head
    user_epilogue<<<4000, 256, 0, stream>>>(C1, user_emb, fu);
    item_epilogue<<<5000, 128, 0, stream>>>(C2, item_emb, Out4, q, a_map_stu, a_map_sk, fi);
    skill_epilogue<<<64, 256, 0, stream>>>(CD, skill_emb, fk);
    head_kernel<<<4096, 128, 0, stream>>>(user, item, q, fu, fi, fk,
                                          W_fuse_s, b_fuse_s, W_fuse_i, b_fuse_i,
                                          W_pred, b_pred, out);
}
